// BiLSTMCRF_8890582303365
// MI455X (gfx1250) — matrix operand order repaired and checked by reading, not compiled, not run on hardware
//
#include <hip/hip_runtime.h>

// Segmented mean-pool (word pooling) as a one-hot WMMA matmul on gfx1250.
// seg_sum[b] = OneHot[128 x 512] @ feats[b][512 x 256], computed with
// v_wmma_f32_16x16x32_f16 using an exact f16 one-hot A and a hi/lo f16
// split of the f32 features (two WMMAs per tile -> ~f32 accuracy).
// Sorted word ids => per 32-char K-chunk only word tiles overlapping
// [wmin,wmax] are computed (wave-uniform skip, EXEC stays all-ones).

typedef __attribute__((ext_vector_type(16))) _Float16 v16h;
typedef __attribute__((ext_vector_type(8)))  float    v8f;

constexpr int L = 512;
constexpr int B = 256;
constexpr int D = 256;
constexpr int W = 128;          // MAX_WORDS
constexpr int KC = 32;          // K-chunk (chars) per WMMA
constexpr int NCHUNK = L / KC;  // 16
constexpr int THREADS = 512;    // 16 wave32 waves

__global__ __launch_bounds__(THREADS)
void word_pool_wmma(const float* __restrict__ char_feats,   // [L,B,D]
                    const int*   __restrict__ word_ids,     // [B,L]
                    const int*   __restrict__ attn_mask,    // [B,L]
                    float* __restrict__ out_feats,          // [W,B,D]
                    float* __restrict__ out_mask)           // [W,B]
{
    __shared__ int s_seg[L];
    __shared__ int s_cnt[W];
    __shared__ int s_wmin[NCHUNK];
    __shared__ int s_wmax[NCHUNK];
    __shared__ int s_charsum;
    __shared__ int s_wordmax;
    __shared__ __align__(32) _Float16 s_hi[D * KC];   // [d][k] layout
    __shared__ __align__(32) _Float16 s_lo[D * KC];

    const int b    = blockIdx.x;
    const int tid  = threadIdx.x;
    const int lane = tid & 31;
    const int wave = tid >> 5;         // 0..15
    const int hi16 = (lane >> 4) & 1;  // which half of the wave
    const int ln   = lane & 15;

    // ---------------- prologue ----------------
    if (tid == 0) { s_charsum = 0; s_wordmax = -1; }
    if (tid < W)       s_cnt[tid]  = 0;
    if (tid < NCHUNK) { s_wmin[tid] = 0x7fffffff; s_wmax[tid] = -1; }
    __syncthreads();

    const int wid = word_ids[(size_t)b * L + tid];   // THREADS == L: tid == l
    const int am  = attn_mask[(size_t)b * L + tid];
    atomicAdd(&s_charsum, am);
    atomicMax(&s_wordmax, wid);
    __syncthreads();

    const int char_num = s_charsum - 2;
    const int word_num = s_wordmax + 1;
    const bool valid   = (tid >= 1) && (tid < 1 + char_num);
    const int  seg     = valid ? wid : W;            // W = overflow bucket
    s_seg[tid] = seg;
    const int ck = tid / KC;
    atomicMin(&s_wmin[ck], seg);
    if (seg < W) {
        atomicMax(&s_wmax[ck], seg);
        atomicAdd(&s_cnt[seg], 1);
    }

    // tile ownership: wave -> word-tile wt (8 tiles), 8 d-tiles each
    const int wt     = wave >> 1;
    const int wbase  = wt * 16;
    const int dtBase = (wave & 1) * 8;

    v8f acc[8] = {};   // 8 d-tiles of 16x16 f32 accumulators

    // ---------------- main loop over K-chunks ----------------
    for (int kc = 0; kc < NCHUNK; ++kc) {
        __syncthreads();   // previous chunk fully consumed
        // stage 32 x 256 chunk: f32 -> (hi,lo) f16 split, [d][k] in LDS
        #pragma unroll
        for (int p = 0; p < 4; ++p) {
            const int idx  = p * THREADS + tid;   // 0..2047
            const int lrow = idx >> 6;            // 0..31
            const int d4   = (idx & 63) * 4;
            const float4 x = *reinterpret_cast<const float4*>(
                &char_feats[((size_t)(kc * KC + lrow) * B + b) * D + d4]);
            const float xs[4] = {x.x, x.y, x.z, x.w};
            #pragma unroll
            for (int q = 0; q < 4; ++q) {
                const _Float16 h = (_Float16)xs[q];
                const _Float16 l = (_Float16)(xs[q] - (float)h);
                s_hi[(d4 + q) * KC + lrow] = h;
                s_lo[(d4 + q) * KC + lrow] = l;
            }
        }
        __syncthreads();

        // wave-uniform skip: no word of this tile occurs in this chunk
        if (wbase > s_wmax[kc] || wbase + 15 < s_wmin[kc]) continue;

        // one-hot A fragment (16x32 f16): a[h] = (seg[K] == wbase+ln)
        const int segbase = kc * KC;
        v16h a;
        #pragma unroll
        for (int h = 0; h < 16; ++h) {
            const int kk = h + 16 * hi16;  // contiguous K = 16*hi16 + h (matches B)
            a[h] = (s_seg[segbase + kk] == wbase + ln)
                       ? (_Float16)1.0f : (_Float16)0.0f;
        }

        #pragma unroll
        for (int j = 0; j < 8; ++j) {
            const int d = (dtBase + j) * 16 + ln;
            // B fragment: halves h -> K = 16*hi16 + h, contiguous in [d][k]
            const v16h bh = *reinterpret_cast<const v16h*>(
                &s_hi[d * KC + hi16 * 16]);
            const v16h bl = *reinterpret_cast<const v16h*>(
                &s_lo[d * KC + hi16 * 16]);
            acc[j] = __builtin_amdgcn_wmma_f32_16x16x32_f16(
                false, a, false, bh, (short)0, acc[j], false, false);
            acc[j] = __builtin_amdgcn_wmma_f32_16x16x32_f16(
                false, a, false, bl, (short)0, acc[j], false, false);
        }
    }

    // ---------------- epilogue: mean + mask + store ----------------
    #pragma unroll
    for (int j = 0; j < 8; ++j) {
        const int d = (dtBase + j) * 16 + ln;
        #pragma unroll
        for (int r = 0; r < 8; ++r) {
            const int w = wbase + r + 8 * hi16;   // C/D layout: M = r + 8*hi
            const int c = s_cnt[w];
            const float cnt = (float)(c < 1 ? 1 : c);
            const float v = (w < word_num) ? acc[j][r] / cnt : 0.0f;
            out_feats[((size_t)w * B + b) * D + d] = v;
        }
    }
    if (tid < W)
        out_mask[(size_t)tid * B + b] = (tid < word_num) ? 1.0f : 0.0f;
}

extern "C" void kernel_launch(void* const* d_in, const int* in_sizes, int n_in,
                              void* d_out, int out_size, void* d_ws, size_t ws_size,
                              hipStream_t stream) {
    (void)in_sizes; (void)n_in; (void)out_size; (void)d_ws; (void)ws_size;
    const float* char_feats = (const float*)d_in[0];   // [512,256,256] f32
    const int*   word_ids   = (const int*)  d_in[1];   // [256,512] i32
    const int*   attn_mask  = (const int*)  d_in[2];   // [256,512] i32
    float* out_feats = (float*)d_out;                  // [128,256,256]
    float* out_mask  = out_feats + (size_t)W * B * D;  // [128,256]

    word_pool_wmma<<<dim3(B), dim3(THREADS), 0, stream>>>(
        char_feats, word_ids, attn_mask, out_feats, out_mask);
}
